// Model_64879775973999
// MI455X (gfx1250) — matrix operand order repaired and checked by reading, not compiled, not run on hardware
//
#include <hip/hip_runtime.h>

typedef __attribute__((ext_vector_type(16))) _Float16 v16h;
typedef __attribute__((ext_vector_type(8)))  _Float16 v8h;
typedef __attribute__((ext_vector_type(8)))  float    v8f;

#define C_IN  512
#define CONV  64
#define DENSE 128
#define OUTC  7

// ---------------------------------------------------------------------------
// WMMA fragment helpers (CDNA5 16x16x32 f16, wave32)
// A 16-bit frag (16xK, M=lane%16): lanes0-15 halves = K{kb+0..7, kb+16..23},
// lanes16-31 halves = K{kb+8..15, kb+24..31}  (ISA 7.12.2)
// B frag from W^T [N][K] row-major: col N=lane%16, same K order as A frag:
// lanes0-15 K{kb+0..7, kb+16..23}, lanes16-31 K{kb+8..15, kb+24..31}.
// D (f32 16x16): VGPR r -> (M = r + 8*(lane/16), N = lane%16)
// ---------------------------------------------------------------------------
__device__ __forceinline__ v16h frag_combine(v8h lo, v8h hi) {
  return __builtin_shufflevector(lo, hi, 0,1,2,3,4,5,6,7,8,9,10,11,12,13,14,15);
}

__device__ __forceinline__ v16h lds_a_frag(const _Float16* a, int ld, int kb) {
  int lane = threadIdx.x & 31;
  int m    = lane & 15;
  int sel  = (lane >> 4) & 1;
  const _Float16* p = a + m * ld + kb + sel * 8;
  v8h lo = *(const v8h*)p;          // ds_load_b128
  v8h hi = *(const v8h*)(p + 16);   // ds_load_b128
  return frag_combine(lo, hi);
}

__device__ __forceinline__ v16h gl_b_frag(const _Float16* __restrict__ wt, int K, int nt, int kb) {
  int lane = threadIdx.x & 31;
  int col  = lane & 15;
  int k0   = kb + ((lane & 16) ? 8 : 0);
  const _Float16* p = wt + (size_t)(nt * 16 + col) * K + k0;
  v8h lo = *(const v8h*)p;
  v8h hi = *(const v8h*)(p + 16);
  return frag_combine(lo, hi);
}

__device__ __forceinline__ v8f wmma16(v16h a, v16h b, v8f c) {
  return __builtin_amdgcn_wmma_f32_16x16x32_f16(false, a, false, b, (short)0, c, false, false);
}

// ---------------------------------------------------------------------------
// K_prep: fold W1 halves (concat[g,g]@W1 == g@(W1[:512]+W1[512:])), transpose
// all weights to [N][K] f16 (pad W6 cols 7->16 with zeros).
// ---------------------------------------------------------------------------
__global__ void k_prep(const float* __restrict__ W1, const float* __restrict__ W2,
                       const float* __restrict__ W3, const float* __restrict__ W4,
                       const float* __restrict__ W5, const float* __restrict__ W6,
                       _Float16* w1t, _Float16* w2t, _Float16* w3t,
                       _Float16* w4t, _Float16* w5t, _Float16* w6t) {
  int i = blockIdx.x * 256 + threadIdx.x;
  if (i < 512 * 512) {
    int n = i / 512, k = i % 512;
    w1t[i] = (_Float16)(W1[(size_t)k * 512 + n] + W1[(size_t)(k + 512) * 512 + n]);
    return;
  }
  i -= 512 * 512;
  if (i < 64 * 512) { int n = i / 512, k = i % 512; w2t[i] = (_Float16)W2[(size_t)k * 64 + n];  return; }
  i -= 64 * 512;
  if (i < 64 * 128) { int n = i / 128, k = i % 128; w3t[i] = (_Float16)W3[(size_t)k * 64 + n];  return; }
  i -= 64 * 128;
  if (i < 128 * 64) { int n = i / 64,  k = i % 64;  w4t[i] = (_Float16)W4[(size_t)k * 128 + n]; return; }
  i -= 128 * 64;
  if (i < 128 * 128){ int n = i / 128, k = i % 128; w5t[i] = (_Float16)W5[(size_t)k * 128 + n]; return; }
  i -= 128 * 128;
  if (i < 16 * 128) {
    int n = i / 128, k = i % 128;
    w6t[i] = (n < OUTC) ? (_Float16)W6[(size_t)k * OUTC + n] : (_Float16)0.0f;
  }
}

// cnt[a] = #rows of M1 with atom a  (= deg(a)+1)
__global__ void k_cnt(const int* __restrict__ m1_atom, int* cnt, int E1) {
  int i = blockIdx.x * 256 + threadIdx.x;
  if (i < E1) atomicAdd(&cnt[m1_atom[i]], 1);
}

// zero T[0 : num_pairs*64]  (num_pairs only known on device)
__global__ void k_zero_t(float* T, const int* __restrict__ np_ptr) {
  long long n = (long long)(*np_ptr) * 64;
  long long stride = (long long)gridDim.x * 256;
  for (long long i = (long long)blockIdx.x * 256 + threadIdx.x; i < n; i += stride)
    T[i] = 0.0f;
}

// ---------------------------------------------------------------------------
// K_y: per 16-node tile, Y = relu(x @ W1eff + b1) @ W2 + b2   -> Y [N,64] f32
// ---------------------------------------------------------------------------
__global__ void __launch_bounds__(256) k_y(const float* __restrict__ x,
    const _Float16* __restrict__ w1t, const float* __restrict__ b1,
    const _Float16* __restrict__ w2t, const float* __restrict__ b2,
    float* __restrict__ Y, int N) {
  __shared__ __align__(16) _Float16 sX[16 * C_IN];
  __shared__ __align__(16) _Float16 sH[16 * C_IN];
  int base = blockIdx.x * 16;
  for (int i = threadIdx.x; i < 16 * C_IN; i += 256) {
    int r = i >> 9, c = i & 511;
    int node = base + r;
    sX[i] = (node < N) ? (_Float16)x[(size_t)node * C_IN + c] : (_Float16)0.0f;
  }
  __syncthreads();
  int wave = threadIdx.x >> 5, lane = threadIdx.x & 31;
  int col0 = lane & 15, mb = ((lane >> 4) & 1) << 3;
  // H = relu(X @ W1eff + b1): 32 col tiles, K=512
  for (int nt = wave; nt < 32; nt += 8) {
    v8f acc = {};
#pragma unroll
    for (int kt = 0; kt < 16; ++kt)
      acc = wmma16(lds_a_frag(sX, C_IN, kt * 32), gl_b_frag(w1t, C_IN, nt, kt * 32), acc);
    int n = nt * 16 + col0;
    float bv = b1[n];
#pragma unroll
    for (int r = 0; r < 8; ++r) {
      float v = acc[r] + bv;
      sH[(mb + r) * C_IN + n] = (_Float16)(v > 0.0f ? v : 0.0f);
    }
  }
  __syncthreads();
  // Y = H @ W2 + b2: 4 col tiles, K=512
  if (wave < 4) {
    v8f acc = {};
#pragma unroll
    for (int kt = 0; kt < 16; ++kt)
      acc = wmma16(lds_a_frag(sH, C_IN, kt * 32), gl_b_frag(w2t, C_IN, wave, kt * 32), acc);
    int n = wave * 16 + col0;
    float bv = b2[n];
#pragma unroll
    for (int r = 0; r < 8; ++r) {
      int node = base + mb + r;
      if (node < N) Y[(size_t)node * CONV + n] = acc[r] + bv;
    }
  }
}

// ---------------------------------------------------------------------------
// K_t: T[o_pair[o]] += Y[m4_atom[o_row[o]]]   (f32 atomics, float4 per thread)
// ---------------------------------------------------------------------------
__global__ void k_t(const float* __restrict__ Y, const int* __restrict__ o_row,
                    const int* __restrict__ o_pair, const int* __restrict__ m4_atom,
                    float* __restrict__ T, long long O) {
  long long idx = (long long)blockIdx.x * 256 + threadIdx.x;
  if (idx >= O * 16) return;
  long long o = idx >> 4;
  int q = (int)(idx & 15);
  int a = m4_atom[o_row[o]];
  int p = o_pair[o];
  float4 v = *(const float4*)(Y + (size_t)a * CONV + q * 4);
  float* t = T + (size_t)p * CONV + q * 4;
  atomicAdd(t + 0, v.x); atomicAdd(t + 1, v.y);
  atomicAdd(t + 2, v.z); atomicAdd(t + 3, v.w);
}

// ---------------------------------------------------------------------------
// K_fused: per 16-row E4 tile: build A=[dest0|dest1] in LDS, run the whole
// W3->W4->W5->W6 chain with WMMA in LDS, atomic-scatter 7 outputs per row.
// ---------------------------------------------------------------------------
__global__ void __launch_bounds__(256) k_fused(
    const float* __restrict__ Y, const float* __restrict__ T,
    const int* __restrict__ cnt, const int* __restrict__ m4_owner,
    const int* __restrict__ m4_atom, const int* __restrict__ o_row,
    const int* __restrict__ o_pair,
    const _Float16* __restrict__ w3t, const float* __restrict__ b3,
    const _Float16* __restrict__ w4t, const float* __restrict__ b4,
    const _Float16* __restrict__ w5t, const float* __restrict__ b5,
    const _Float16* __restrict__ w6t, const float* __restrict__ b6,
    float* __restrict__ out, int E4, long long O) {
  __shared__ __align__(16) _Float16 sA [16 * (2 * CONV)];
  __shared__ __align__(16) _Float16 sH3[16 * CONV];
  __shared__ __align__(16) _Float16 sH4[16 * DENSE];
  __shared__ __align__(16) _Float16 sH5[16 * DENSE];
  __shared__ int sBound[17];
  int base = blockIdx.x * 16;
  // row ranges in sorted o_row via binary search
  if (threadIdx.x < 17) {
    int e = base + (int)threadIdx.x; if (e > E4) e = E4;
    long long lo = 0, hi = O;
    while (lo < hi) { long long mid = (lo + hi) >> 1; if (o_row[mid] < e) lo = mid + 1; else hi = mid; }
    sBound[threadIdx.x] = (int)lo;
  }
  __syncthreads();
  // build A: cols 0-63 = dest0 (sum of T rows), cols 64-127 = dest1 = cnt*Y
  {
    int g = threadIdx.x >> 4, c4 = threadIdx.x & 15;
    int e = base + g;
    float a0x = 0, a0y = 0, a0z = 0, a0w = 0;
    float d1x = 0, d1y = 0, d1z = 0, d1w = 0;
    if (e < E4) {
      int a = m4_atom[e];
      float cv = (float)cnt[a];
      float4 yv = *(const float4*)(Y + (size_t)a * CONV + c4 * 4);
      d1x = cv * yv.x; d1y = cv * yv.y; d1z = cv * yv.z; d1w = cv * yv.w;
      int s = sBound[g], t = sBound[g + 1];
      for (int o = s; o < t; ++o) {
        float4 tv = *(const float4*)(T + (size_t)o_pair[o] * CONV + c4 * 4);
        a0x += tv.x; a0y += tv.y; a0z += tv.z; a0w += tv.w;
      }
    }
    int cb = c4 * 4;
    sA[g * 128 + cb + 0] = (_Float16)a0x; sA[g * 128 + cb + 1] = (_Float16)a0y;
    sA[g * 128 + cb + 2] = (_Float16)a0z; sA[g * 128 + cb + 3] = (_Float16)a0w;
    sA[g * 128 + 64 + cb + 0] = (_Float16)d1x; sA[g * 128 + 64 + cb + 1] = (_Float16)d1y;
    sA[g * 128 + 64 + cb + 2] = (_Float16)d1z; sA[g * 128 + 64 + cb + 3] = (_Float16)d1w;
  }
  __syncthreads();
  int wave = threadIdx.x >> 5, lane = threadIdx.x & 31;
  int col0 = lane & 15, mb = ((lane >> 4) & 1) << 3;
  // h3 = relu(A @ W3 + b3): 4 col tiles, K=128
  if (wave < 4) {
    v8f acc = {};
#pragma unroll
    for (int kt = 0; kt < 4; ++kt)
      acc = wmma16(lds_a_frag(sA, 128, kt * 32), gl_b_frag(w3t, 128, wave, kt * 32), acc);
    int n = wave * 16 + col0;
    float bv = b3[n];
#pragma unroll
    for (int r = 0; r < 8; ++r) {
      float v = acc[r] + bv;
      sH3[(mb + r) * CONV + n] = (_Float16)(v > 0.0f ? v : 0.0f);
    }
  }
  __syncthreads();
  // h4 = h3 @ W4 + b4: 8 col tiles, K=64
  {
    v8f acc = {};
#pragma unroll
    for (int kt = 0; kt < 2; ++kt)
      acc = wmma16(lds_a_frag(sH3, CONV, kt * 32), gl_b_frag(w4t, CONV, wave, kt * 32), acc);
    int n = wave * 16 + col0;
    float bv = b4[n];
#pragma unroll
    for (int r = 0; r < 8; ++r)
      sH4[(mb + r) * DENSE + n] = (_Float16)(acc[r] + bv);
  }
  __syncthreads();
  // h5 = relu(h4 @ W5 + b5): 8 col tiles, K=128
  {
    v8f acc = {};
#pragma unroll
    for (int kt = 0; kt < 4; ++kt)
      acc = wmma16(lds_a_frag(sH4, DENSE, kt * 32), gl_b_frag(w5t, DENSE, wave, kt * 32), acc);
    int n = wave * 16 + col0;
    float bv = b5[n];
#pragma unroll
    for (int r = 0; r < 8; ++r) {
      float v = acc[r] + bv;
      sH5[(mb + r) * DENSE + n] = (_Float16)(v > 0.0f ? v : 0.0f);
    }
  }
  __syncthreads();
  // h6 = h5 @ W6pad + b6 (1 tile, 7 valid cols), scatter to out[m4_owner]
  if (wave == 0) {
    v8f acc = {};
#pragma unroll
    for (int kt = 0; kt < 4; ++kt)
      acc = wmma16(lds_a_frag(sH5, DENSE, kt * 32), gl_b_frag(w6t, DENSE, 0, kt * 32), acc);
    if (col0 < OUTC) {
      float bv = b6[col0];
#pragma unroll
      for (int r = 0; r < 8; ++r) {
        int e = base + mb + r;
        if (e < E4) atomicAdd(&out[(size_t)m4_owner[e] * OUTC + col0], acc[r] + bv);
      }
    }
  }
}

// ---------------------------------------------------------------------------
extern "C" void kernel_launch(void* const* d_in, const int* in_sizes, int n_in,
                              void* d_out, int out_size, void* d_ws, size_t ws_size,
                              hipStream_t stream) {
  const float* x  = (const float*)d_in[0];
  const float* W1 = (const float*)d_in[1];  const float* b1 = (const float*)d_in[2];
  const float* W2 = (const float*)d_in[3];  const float* b2 = (const float*)d_in[4];
  const float* W3 = (const float*)d_in[5];  const float* b3 = (const float*)d_in[6];
  const float* W4 = (const float*)d_in[7];  const float* b4 = (const float*)d_in[8];
  const float* W5 = (const float*)d_in[9];  const float* b5 = (const float*)d_in[10];
  const float* W6 = (const float*)d_in[11]; const float* b6 = (const float*)d_in[12];
  const int* m1_atom  = (const int*)d_in[13];
  const int* m4_owner = (const int*)d_in[14];
  const int* m4_atom  = (const int*)d_in[15];
  const int* o_row    = (const int*)d_in[16];
  const int* o_pair   = (const int*)d_in[17];
  const int* np_dev   = (const int*)d_in[19];

  int E1 = in_sizes[13];
  int E4 = in_sizes[14];
  long long O = in_sizes[16];
  int N = in_sizes[0] / C_IN;

  char* ws = (char*)d_ws;
  size_t off = 0;
  auto carve = [&](size_t bytes) { void* p = ws + off; off += (bytes + 255) & ~(size_t)255; return p; };
  _Float16* w1t = (_Float16*)carve((size_t)C_IN * C_IN * 2);
  _Float16* w2t = (_Float16*)carve((size_t)CONV * C_IN * 2);
  _Float16* w3t = (_Float16*)carve((size_t)CONV * 128 * 2);
  _Float16* w4t = (_Float16*)carve((size_t)DENSE * CONV * 2);
  _Float16* w5t = (_Float16*)carve((size_t)DENSE * DENSE * 2);
  _Float16* w6t = (_Float16*)carve((size_t)16 * DENSE * 2);
  int*   cnt = (int*)carve((size_t)N * 4);
  float* Y   = (float*)carve((size_t)N * CONV * 4);
  float* T   = (float*)carve((size_t)O * CONV * 4);  // num_pairs <= O

  hipMemsetAsync(cnt, 0, (size_t)N * 4, stream);
  hipMemsetAsync(d_out, 0, (size_t)N * OUTC * 4, stream);

  const int prep_total = 512*512 + 64*512 + 64*128 + 128*64 + 128*128 + 16*128;
  k_prep<<<(prep_total + 255) / 256, 256, 0, stream>>>(W1, W2, W3, W4, W5, W6,
                                                       w1t, w2t, w3t, w4t, w5t, w6t);
  k_zero_t<<<4096, 256, 0, stream>>>(T, np_dev);
  if (E1 > 0) k_cnt<<<(E1 + 255) / 256, 256, 0, stream>>>(m1_atom, cnt, E1);
  k_y<<<(N + 15) / 16, 256, 0, stream>>>(x, w1t, b1, w2t, b2, Y, N);
  if (O > 0)
    k_t<<<(unsigned)((O * 16 + 255) / 256), 256, 0, stream>>>(Y, o_row, o_pair, m4_atom, T, O);
  if (E4 > 0)
    k_fused<<<(E4 + 15) / 16, 256, 0, stream>>>(Y, T, cnt, m4_owner, m4_atom, o_row, o_pair,
                                                w3t, b3, w4t, b4, w5t, b5, w6t, b6,
                                                (float*)d_out, E4, O);
}
